// Attention_25769803776363
// MI455X (gfx1250) — matrix operand order repaired and checked by reading, not compiled, not run on hardware
//
#include <hip/hip_runtime.h>

// ---------------------------------------------------------------------------
// MI455X (gfx1250) multi-head attention:
//   qkv = x @ W_qkv + b_qkv  ->  split k,q,v (k FIRST per reference)
//   flash-style softmax(q k^T / sqrt(64)) @ v  ->  @ W_out + b_out
// Matmuls via v_wmma_f32_16x16x32_bf16 (bf16 in, f32 accumulate).
// Attention K/V blocks staged once per workgroup via async Global->LDS DMA
// (global_load_async_to_lds_b128 + s_wait_asynccnt), consumed as ds_load_b128.
// ---------------------------------------------------------------------------

typedef __bf16 bf16_t;
typedef __attribute__((ext_vector_type(16))) __bf16 v16bf;
typedef __attribute__((ext_vector_type(8)))  float  v8f;

#define BATCH   2
#define SEQ     2048
#define EMB     1024
#define HEADS   16
#define HD      64
#define QKVCOLS 3072
#define ROWS    (BATCH * SEQ)   // 4096

union FragAB {            // one wave-lane's share of a 16x32 bf16 fragment
  v16bf v;
  uint4 u[2];
};

__device__ __forceinline__ v8f wmma_bf16(v16bf a, v16bf b, v8f c) {
  // (neg_a, A, neg_b, B, c_mod, C, reuse_a, reuse_b)
  return __builtin_amdgcn_wmma_f32_16x16x32_bf16(false, a, false, b,
                                                 (short)0, c, false, false);
}

// A fragment (16x32 bf16) from row-major [.. x ldk] bf16; row_base -> [row0][k0].
// ISA 7.12.2: lanes 0-15 hold K 0-7 & 16-23 of row=lane; lanes 16-31 K 8-15 & 24-31.
__device__ __forceinline__ v16bf load_a_frag(const bf16_t* row_base, int ldk, int lane) {
  int r    = lane & 15;
  int half = lane >> 4;
  const bf16_t* p = row_base + (size_t)r * ldk + half * 8;
  FragAB f;
  f.u[0] = *(const uint4*)(p);        // K = half*8 + 0..7
  f.u[1] = *(const uint4*)(p + 16);   // K = 16 + half*8 + 0..7
  return f.v;
}

// B fragment (32x16 bf16) from row-major B^T storage [N x K]; col_base -> [col0][k0].
// Lanes 0-15: K 0-7 & 16-23 of col=lane; lanes 16-31: K 8-15 & 24-31.
__device__ __forceinline__ v16bf load_b_frag(const bf16_t* col_base, int ldk, int lane) {
  int c    = lane & 15;
  int half = lane >> 4;
  const bf16_t* p = col_base + (size_t)c * ldk + half * 8;
  FragAB f;
  f.u[0] = *(const uint4*)(p);        // K = half*8 + 0..7
  f.u[1] = *(const uint4*)(p + 16);   // K = 16 + half*8 + 0..7
  return f.v;
}

// Async DMA: copy 16 bytes global -> LDS, tracked on ASYNCcnt (per-lane).
// VDST supplies the LDS byte offset.
__device__ __forceinline__ void async_copy16(unsigned lds_off, const bf16_t* gsrc) {
  unsigned long long ga = (unsigned long long)(size_t)gsrc;
  asm volatile("global_load_async_to_lds_b128 %0, %1, off"
               :: "v"(lds_off), "v"(ga) : "memory");
}

__device__ __forceinline__ void wait_async0() {
  asm volatile("s_wait_asynccnt 0x0" ::: "memory");
}

// ---------------------------------------------------------------------------
// Conversions (one-shot, bandwidth-trivial; weights live in L2 afterwards)
// ---------------------------------------------------------------------------
__global__ void cvt_bf16_kernel(const float* __restrict__ in,
                                bf16_t* __restrict__ out, int n) {
  int i = blockIdx.x * blockDim.x + threadIdx.x;
  if (i < n) out[i] = (bf16_t)in[i];
}

// in: [K][N] f32  ->  out: [N][K] bf16 (coalesced writes; reads hit L2)
__global__ void cvt_transpose_kernel(const float* __restrict__ in,
                                     bf16_t* __restrict__ out, int K, int N) {
  int k = blockIdx.x * blockDim.x + threadIdx.x;
  int c = blockIdx.y;
  if (k < K) out[(size_t)c * K + k] = (bf16_t)in[(size_t)k * N + c];
}

// ---------------------------------------------------------------------------
// QKV GEMM: [4096 x 1024] @ [1024 x 3072] + bias, epilogue scatters into
//   Kh [B,H,N,64]  (cols 0..1023)
//   Qh [B,H,N,64]  (cols 1024..2047, pre-scaled by 1/sqrt(64) = 0.125)
//   Vt [B,H,64,N]  (cols 2048..3071, transposed for PV B-fragments)
// Block: 256 threads (8 waves), 128x128 tile; wave: 32x64 (8 WMMA / K-step).
// All 6 fragments of a K-step are preloaded before the WMMA burst so the
// scheduler can stage s_wait_loadcnt and keep the XDL pipe busy.
// ---------------------------------------------------------------------------
__global__ __launch_bounds__(256)
void qkv_gemm_kernel(const bf16_t* __restrict__ Xb,   // [4096][1024]
                     const bf16_t* __restrict__ Wt,   // [3072][1024] = W_qkv^T
                     const float*  __restrict__ bias, // [3072]
                     bf16_t* __restrict__ Qh,
                     bf16_t* __restrict__ Kh,
                     bf16_t* __restrict__ Vt) {
  const int K = EMB;
  int lane = threadIdx.x & 31;
  int wave = threadIdx.x >> 5;
  int rowBase = blockIdx.x * 128 + (wave >> 1) * 32;
  int colBase = blockIdx.y * 128 + (wave & 1) * 64;

  v8f zero = {};
  v8f acc[2][4];
  #pragma unroll
  for (int i = 0; i < 2; i++)
    #pragma unroll
    for (int j = 0; j < 4; j++) acc[i][j] = zero;

  for (int k0 = 0; k0 < K; k0 += 32) {
    if (k0 + 32 < K) {  // hint next K-tile into cache (global_prefetch_b8)
      __builtin_prefetch(Xb + (size_t)rowBase * K + k0 + 32, 0, 1);
      __builtin_prefetch(Wt + (size_t)colBase * K + k0 + 32, 0, 1);
    }
    v16bf a[2], b[4];
    #pragma unroll
    for (int i = 0; i < 2; i++)
      a[i] = load_a_frag(Xb + (size_t)(rowBase + i * 16) * K + k0, K, lane);
    #pragma unroll
    for (int j = 0; j < 4; j++)
      b[j] = load_b_frag(Wt + (size_t)(colBase + j * 16) * K + k0, K, lane);
    #pragma unroll
    for (int j = 0; j < 4; j++)
      #pragma unroll
      for (int i = 0; i < 2; i++)
        acc[i][j] = wmma_bf16(a[i], b[j], acc[i][j]);
  }

  // C-fragment: reg r = row (r | r+8 by lane half), lane&15 = col
  int cll = lane & 15;
  int rhi = (lane >> 4) * 8;
  #pragma unroll
  for (int i = 0; i < 2; i++) {
    #pragma unroll
    for (int j = 0; j < 4; j++) {
      int colg = colBase + j * 16 + cll;
      float bv = bias[colg];
      #pragma unroll
      for (int r = 0; r < 8; r++) {
        int rowg = rowBase + i * 16 + rhi + r;
        float v  = acc[i][j][r] + bv;
        int b_ = rowg >> 11;      // / SEQ
        int n_ = rowg & (SEQ - 1);
        if (colg < 1024) {                         // k  (first in split!)
          int h = colg >> 6, d = colg & 63;
          Kh[((size_t)(b_ * HEADS + h) * SEQ + n_) * HD + d] = (bf16_t)v;
        } else if (colg < 2048) {                  // q, pre-scaled
          int c2 = colg - 1024;
          int h = c2 >> 6, d = c2 & 63;
          Qh[((size_t)(b_ * HEADS + h) * SEQ + n_) * HD + d] = (bf16_t)(v * 0.125f);
        } else {                                   // v, transposed
          int c2 = colg - 2048;
          int h = c2 >> 6, d = c2 & 63;
          Vt[((size_t)(b_ * HEADS + h) * HD + d) * SEQ + n_] = (bf16_t)v;
        }
      }
    }
  }
}

// ---------------------------------------------------------------------------
// Flash attention: block = 4 waves x 16 query rows; stream 64-key blocks.
// K/V blocks (8KB each) are DMA'd once per block into LDS (async-to-LDS),
// then all 4 waves consume them as B-fragments via ds_load_b128.
// Per block-iter per wave: 8 WMMA (QK^T) + 8 WMMA (PV), online softmax.
// ---------------------------------------------------------------------------
__global__ __launch_bounds__(128)
void attn_kernel(const bf16_t* __restrict__ Qh,
                 const bf16_t* __restrict__ Kh,
                 const bf16_t* __restrict__ Vt,
                 bf16_t* __restrict__ AttnOut) {   // [4096][1024] bf16
  __shared__ __align__(16) bf16_t Ksh[64 * 64];       // [key][d]
  __shared__ __align__(16) bf16_t Vsh[64 * 64];       // [d][key]  (V^T slice)
  __shared__ __align__(16) bf16_t Plds[4][16 * 64];

  int tid  = threadIdx.x;
  int lane = tid & 31;
  int wave = tid >> 5;
  int bh   = blockIdx.y;            // 0..31
  int b_   = bh >> 4;
  int h    = bh & 15;
  int qn   = blockIdx.x * 64 + wave * 16;

  const bf16_t* Qbase = Qh + ((size_t)bh * SEQ + qn) * HD;
  const bf16_t* Kbase = Kh + (size_t)bh * SEQ * HD;
  const bf16_t* Vbase = Vt + (size_t)bh * HD * SEQ;

  unsigned kshBase = (unsigned)(size_t)&Ksh[0];   // LDS byte offsets
  unsigned vshBase = (unsigned)(size_t)&Vsh[0];

  v16bf qa[2];
  #pragma unroll
  for (int kk = 0; kk < 2; kk++)
    qa[kk] = load_a_frag(Qbase + kk * 32, HD, lane);

  v8f zero = {};
  v8f o[4];
  #pragma unroll
  for (int j = 0; j < 4; j++) o[j] = zero;
  float mrow[8], lrow[8];
  #pragma unroll
  for (int r = 0; r < 8; r++) { mrow[r] = -1e30f; lrow[r] = 0.f; }

  for (int nb = 0; nb < SEQ; nb += 64) {
    // ---- async DMA this K/V block into LDS (512 x 16B chunks each) ----
    #pragma unroll
    for (int i = 0; i < 4; i++) {
      int chunk = tid + i * 128;            // 0..511
      int row   = chunk >> 3;               // 0..63
      int c16   = (chunk & 7) * 8;          // element offset within row
      async_copy16(kshBase + (unsigned)(row * 64 + c16) * 2,
                   Kbase + (size_t)(nb + row) * HD + c16);
      async_copy16(vshBase + (unsigned)(row * 64 + c16) * 2,
                   Vbase + (size_t)row * SEQ + nb + c16);
    }
    wait_async0();
    __syncthreads();

    // ---- S = Q K^T for this 16x64 tile (Q already scaled) ----
    v8f s[4];
    #pragma unroll
    for (int cs = 0; cs < 4; cs++) {
      v8f a0 = zero;
      #pragma unroll
      for (int kk = 0; kk < 2; kk++) {
        v16bf kb = load_b_frag(&Ksh[(cs * 16) * 64 + kk * 32], 64, lane);
        a0 = wmma_bf16(qa[kk], kb, a0);
      }
      s[cs] = a0;
    }
    // ---- online softmax (row = reg index within 16-lane half) ----
    float bm[8];
    #pragma unroll
    for (int r = 0; r < 8; r++)
      bm[r] = fmaxf(fmaxf(s[0][r], s[1][r]), fmaxf(s[2][r], s[3][r]));
    #pragma unroll
    for (int m = 1; m < 16; m <<= 1)
      #pragma unroll
      for (int r = 0; r < 8; r++)
        bm[r] = fmaxf(bm[r], __shfl_xor(bm[r], m, 32));
    float alpha[8];
    #pragma unroll
    for (int r = 0; r < 8; r++) {
      float mn = fmaxf(mrow[r], bm[r]);
      alpha[r] = __expf(mrow[r] - mn);
      mrow[r] = mn;
    }
    float ps[8];
    #pragma unroll
    for (int r = 0; r < 8; r++) ps[r] = 0.f;
    #pragma unroll
    for (int cs = 0; cs < 4; cs++)
      #pragma unroll
      for (int r = 0; r < 8; r++) {
        float p = __expf(s[cs][r] - mrow[r]);
        s[cs][r] = p;
        ps[r] += p;
      }
    #pragma unroll
    for (int m = 1; m < 16; m <<= 1)
      #pragma unroll
      for (int r = 0; r < 8; r++)
        ps[r] += __shfl_xor(ps[r], m, 32);
    #pragma unroll
    for (int r = 0; r < 8; r++) lrow[r] = lrow[r] * alpha[r] + ps[r];
    #pragma unroll
    for (int cs = 0; cs < 4; cs++)
      #pragma unroll
      for (int r = 0; r < 8; r++) o[cs][r] *= alpha[r];

    // ---- relayout P: C-fragment -> row-major bf16 in per-wave LDS ----
    #pragma unroll
    for (int cs = 0; cs < 4; cs++)
      #pragma unroll
      for (int r = 0; r < 8; r++) {
        int row = (lane >> 4) * 8 + r;
        int col = cs * 16 + (lane & 15);
        Plds[wave][row * 64 + col] = (bf16_t)s[cs][r];
      }
    __syncthreads();

    // ---- O += P @ V  (A-frags from LDS, B-frags from staged V^T) ----
    #pragma unroll
    for (int kk = 0; kk < 2; kk++) {
      int rr = lane & 15, half = lane >> 4;
      const bf16_t* p = &Plds[wave][rr * 64 + kk * 32 + half * 8];
      FragAB pf;
      pf.u[0] = *(const uint4*)(p);
      pf.u[1] = *(const uint4*)(p + 16);
      #pragma unroll
      for (int d4 = 0; d4 < 4; d4++) {
        v16bf vb = load_b_frag(&Vsh[(d4 * 16) * 64 + kk * 32], 64, lane);
        o[d4] = wmma_bf16(pf.v, vb, o[d4]);
      }
    }
    __syncthreads();   // protects Ksh/Vsh/Plds against next iteration's writes
  }

  // ---- normalize + store bf16 [B*N][H*64] for the out-projection ----
  #pragma unroll
  for (int d4 = 0; d4 < 4; d4++)
    #pragma unroll
    for (int r = 0; r < 8; r++) {
      int row = (lane >> 4) * 8 + r;
      int col = d4 * 16 + (lane & 15);
      size_t orow = (size_t)b_ * SEQ + (qn + row);
      AttnOut[orow * EMB + h * HD + col] = (bf16_t)(o[d4][r] / lrow[r]);
    }
}

// ---------------------------------------------------------------------------
// Output projection: [4096 x 1024] @ [1024 x 1024] + b_out -> f32 d_out
// ---------------------------------------------------------------------------
__global__ __launch_bounds__(256)
void out_gemm_kernel(const bf16_t* __restrict__ Ab,   // [4096][1024]
                     const bf16_t* __restrict__ Wt,   // [1024][1024] = W_out^T
                     const float*  __restrict__ bias, // [1024]
                     float* __restrict__ Out) {       // [4096][1024]
  const int K = EMB;
  int lane = threadIdx.x & 31;
  int wave = threadIdx.x >> 5;
  int rowBase = blockIdx.x * 128 + (wave >> 1) * 32;
  int colBase = blockIdx.y * 128 + (wave & 1) * 64;

  v8f zero = {};
  v8f acc[2][4];
  #pragma unroll
  for (int i = 0; i < 2; i++)
    #pragma unroll
    for (int j = 0; j < 4; j++) acc[i][j] = zero;

  for (int k0 = 0; k0 < K; k0 += 32) {
    if (k0 + 32 < K) {
      __builtin_prefetch(Ab + (size_t)rowBase * K + k0 + 32, 0, 1);
      __builtin_prefetch(Wt + (size_t)colBase * K + k0 + 32, 0, 1);
    }
    v16bf a[2], b[4];
    #pragma unroll
    for (int i = 0; i < 2; i++)
      a[i] = load_a_frag(Ab + (size_t)(rowBase + i * 16) * K + k0, K, lane);
    #pragma unroll
    for (int j = 0; j < 4; j++)
      b[j] = load_b_frag(Wt + (size_t)(colBase + j * 16) * K + k0, K, lane);
    #pragma unroll
    for (int j = 0; j < 4; j++)
      #pragma unroll
      for (int i = 0; i < 2; i++)
        acc[i][j] = wmma_bf16(a[i], b[j], acc[i][j]);
  }

  int cll = lane & 15;
  int rhi = (lane >> 4) * 8;
  #pragma unroll
  for (int i = 0; i < 2; i++)
    #pragma unroll
    for (int j = 0; j < 4; j++) {
      int colg = colBase + j * 16 + cll;
      float bv = bias[colg];
      #pragma unroll
      for (int r = 0; r < 8; r++) {
        int rowg = rowBase + i * 16 + rhi + r;
        Out[(size_t)rowg * EMB + colg] = acc[i][j][r] + bv;
      }
    }
}

// ---------------------------------------------------------------------------
// Workspace layout (48 MB total):
//  [ 0,  8M) x bf16         [ 8M, 14M) W_qkv^T bf16   [14M, 16M) W_out^T bf16
//  [16M, 24M) Q heads bf16  [24M, 32M) K heads bf16   [32M, 40M) V^T bf16
//  [40M, 48M) attn-out bf16
// ---------------------------------------------------------------------------
extern "C" void kernel_launch(void* const* d_in, const int* in_sizes, int n_in,
                              void* d_out, int out_size, void* d_ws, size_t ws_size,
                              hipStream_t stream) {
  const float* x    = (const float*)d_in[0];
  const float* Wqkv = (const float*)d_in[1];
  const float* bqkv = (const float*)d_in[2];
  const float* Wout = (const float*)d_in[3];
  const float* bout = (const float*)d_in[4];
  float* out = (float*)d_out;

  char* ws = (char*)d_ws;
  const size_t MB = 1024u * 1024u;
  bf16_t* Xb   = (bf16_t*)(ws);
  bf16_t* Wqt  = (bf16_t*)(ws +  8 * MB);
  bf16_t* Wot  = (bf16_t*)(ws + 14 * MB);
  bf16_t* Qh   = (bf16_t*)(ws + 16 * MB);
  bf16_t* Kh   = (bf16_t*)(ws + 24 * MB);
  bf16_t* Vt   = (bf16_t*)(ws + 32 * MB);
  bf16_t* Attn = (bf16_t*)(ws + 40 * MB);

  cvt_bf16_kernel<<<(ROWS * EMB) / 256, 256, 0, stream>>>(x, Xb, ROWS * EMB);
  cvt_transpose_kernel<<<dim3(EMB / 256, QKVCOLS), 256, 0, stream>>>(Wqkv, Wqt, EMB, QKVCOLS);
  cvt_transpose_kernel<<<dim3(EMB / 256, EMB), 256, 0, stream>>>(Wout, Wot, EMB, EMB);

  qkv_gemm_kernel<<<dim3(ROWS / 128, QKVCOLS / 128), 256, 0, stream>>>(
      Xb, Wqt, bqkv, Qh, Kh, Vt);

  attn_kernel<<<dim3(SEQ / 64, BATCH * HEADS), 128, 0, stream>>>(Qh, Kh, Vt, Attn);

  out_gemm_kernel<<<dim3(ROWS / 128, EMB / 128), 256, 0, stream>>>(Attn, Wot, bout, out);
}
